// FAGNN_63694365000313
// MI455X (gfx1250) — matrix operand order repaired and checked by reading, not compiled, not run on hardware
//
#include <hip/hip_runtime.h>
#include <hip/hip_bf16.h>

typedef __attribute__((ext_vector_type(16))) _Float16 v16h;
typedef __attribute__((ext_vector_type(8)))  float    v8f;

#define NN 512
#define EE 32

// ---------------------------------------------------------------------------
// K1: frame setup — center, 3x3 covariance, Jacobi eigh, build F_ops[8][3][3]
// ---------------------------------------------------------------------------
__global__ void frame_kernel(const float* __restrict__ x,
                             float* __restrict__ F,      // 72 floats
                             float* __restrict__ center) // 3 floats
{
    if (threadIdx.x != 0 || blockIdx.x != 0) return;
    float c0 = 0.f, c1 = 0.f, c2 = 0.f;
    for (int n = 0; n < NN; ++n) { c0 += x[n*3+0]; c1 += x[n*3+1]; c2 += x[n*3+2]; }
    c0 *= (1.f/NN); c1 *= (1.f/NN); c2 *= (1.f/NN);
    center[0] = c0; center[1] = c1; center[2] = c2;

    float A[3][3] = {{0,0,0},{0,0,0},{0,0,0}};
    for (int n = 0; n < NN; ++n) {
        float a = x[n*3+0]-c0, b = x[n*3+1]-c1, d = x[n*3+2]-c2;
        A[0][0] += a*a; A[0][1] += a*b; A[0][2] += a*d;
        A[1][1] += b*b; A[1][2] += b*d; A[2][2] += d*d;
    }
    A[1][0] = A[0][1]; A[2][0] = A[0][2]; A[2][1] = A[1][2];

    float V[3][3] = {{1,0,0},{0,1,0},{0,0,1}};
    const int PQ[3][2] = {{0,1},{0,2},{1,2}};
    for (int sweep = 0; sweep < 20; ++sweep) {
        for (int r = 0; r < 3; ++r) {
            int p = PQ[r][0], q = PQ[r][1];
            float apq = A[p][q];
            if (fabsf(apq) < 1e-24f) continue;
            float theta = (A[q][q] - A[p][p]) / (2.f * apq);
            float t = ((theta >= 0.f) ? 1.f : -1.f) /
                      (fabsf(theta) + sqrtf(theta*theta + 1.f));
            float cc = 1.f / sqrtf(t*t + 1.f);
            float ss = t * cc;
            for (int k = 0; k < 3; ++k) { // A <- A*G
                float akp = A[k][p], akq = A[k][q];
                A[k][p] = cc*akp - ss*akq; A[k][q] = ss*akp + cc*akq;
            }
            for (int k = 0; k < 3; ++k) { // A <- G^T*A
                float apk = A[p][k], aqk = A[q][k];
                A[p][k] = cc*apk - ss*aqk; A[q][k] = ss*apk + cc*aqk;
            }
            for (int k = 0; k < 3; ++k) { // V <- V*G
                float vkp = V[k][p], vkq = V[k][q];
                V[k][p] = cc*vkp - ss*vkq; V[k][q] = ss*vkp + cc*vkq;
            }
        }
    }
    float w[3] = { A[0][0], A[1][1], A[2][2] };
    int ord[3] = {0,1,2};
    if (w[ord[0]] > w[ord[1]]) { int t = ord[0]; ord[0] = ord[1]; ord[1] = t; }
    if (w[ord[1]] > w[ord[2]]) { int t = ord[1]; ord[1] = ord[2]; ord[2] = t; }
    if (w[ord[0]] > w[ord[1]]) { int t = ord[0]; ord[0] = ord[1]; ord[1] = t; }

    for (int o = 0; o < 8; ++o)
        for (int i = 0; i < 3; ++i)
            for (int j = 0; j < 3; ++j) {
                float s = ((o >> (2 - j)) & 1) ? -1.f : 1.f;
                F[o*9 + i*3 + j] = s * V[i][ord[j]];
            }
}

// ---------------------------------------------------------------------------
// K2: embed — h0[f][n][0:3]=F^T xc, [3:6]=F^T v, [6:14]=node_feat (stride 16)
// ---------------------------------------------------------------------------
__global__ void embed_kernel(const float* __restrict__ x, const float* __restrict__ v,
                             const float* __restrict__ nf, const float* __restrict__ F,
                             const float* __restrict__ center, float* __restrict__ h0)
{
    int tid = blockIdx.x * blockDim.x + threadIdx.x; // 8*512
    int f = tid >> 9, n = tid & 511;
    float xc[3], vv[3];
    for (int j = 0; j < 3; ++j) { xc[j] = x[n*3+j] - center[j]; vv[j] = v[n*3+j]; }
    const float* Ff = F + f*9;
    float* hp = h0 + tid * 16;
    for (int i = 0; i < 3; ++i) {
        hp[i]   = Ff[0+i]*xc[0] + Ff[3+i]*xc[1] + Ff[6+i]*xc[2]; // sum_j F[j][i]*xc[j]
        hp[3+i] = Ff[0+i]*vv[0] + Ff[3+i]*vv[1] + Ff[6+i]*vv[2];
    }
    for (int k = 0; k < 8; ++k) hp[6+k] = nf[n*8+k];
    hp[14] = 0.f; hp[15] = 0.f;
}

// ---------------------------------------------------------------------------
// K3: prep — per-node projections pj = h@W1_j, pi = h@W1_i (one wave per f,n)
// ---------------------------------------------------------------------------
__global__ void prep_kernel(const float* __restrict__ h, int dh, int stride_h,
                            const float* __restrict__ w1j, const float* __restrict__ w1i,
                            float* __restrict__ pj, float* __restrict__ pi)
{
    int gw = (blockIdx.x * blockDim.x + threadIdx.x) >> 5; // f*512+n
    int lane = threadIdx.x & 31;
    const float* hp = h + gw * stride_h;
    float aj = 0.f, ai = 0.f;
    for (int k = 0; k < dh; ++k) {
        float hk = hp[k];
        aj = fmaf(hk, w1j[k*EE + lane], aj);
        ai = fmaf(hk, w1i[k*EE + lane], ai);
    }
    pj[gw*EE + lane] = aj;
    pi[gw*EE + lane] = ai;
}

// ---------------------------------------------------------------------------
// K4: edge MLP + neighbor sum — WMMA workhorse. No LDS, no cross-lane ops in
// the hot loop: lane l computes t-row (l&15) directly for the K-columns its
// A-fragment needs, K = {kh..kh+7, kh+16..kh+23}, kh = (l>>4)*8. The edge
// bias b2 rides in the WMMA C operand. Then agg[i] += sum_j relu(t@W2 + b2).
// ---------------------------------------------------------------------------
__global__ void __launch_bounds__(256)
edge_kernel(const float* __restrict__ ef, const float* __restrict__ pj,
            const float* __restrict__ pi, const float* __restrict__ w1e,
            const float* __restrict__ b1, const float* __restrict__ W2,
            const float* __restrict__ b2, float* __restrict__ agg)
{
    const int lane = threadIdx.x & 31;
    const int w    = threadIdx.x >> 5;
    const int f    = blockIdx.y;
    const int i    = blockIdx.x * 8 + w;
    const int col  = lane & 15;       // t-row this lane computes; also its D column
    const int kh   = (lane >> 4) * 8; // A-fragment K base: {kh..kh+7, kh+16..kh+23}

    // Gather B tiles of W2 (K=32 x N=16 each) into ISA layout, f16:
// lane n holds column n; lanes 0-15 hold K={0-7,16-23}, lanes 16-31 K={8-15,24-31}.
    v16h b0, b1v;
    for (int m = 0; m < 16; ++m) {
        int k = (lane >> 4) * 8 + ((m < 8) ? m : m + 8);
        b0[m]  = (_Float16)W2[k*EE + col];
        b1v[m] = (_Float16)W2[k*EE + 16 + col];
    }

    // Per-lane column constants, in A-fragment element order.
    float w1ev[16], pibv[16];
    for (int c = 0; c < 16; ++c) {
        int cc = kh + ((c < 8) ? c : c + 8);
        w1ev[c] = w1e[cc];
        pibv[c] = pi[(size_t)(f*NN + i)*EE + cc] + b1[cc];
    }
    // Edge bias folded into the WMMA accumulator input.
    const float b2a = b2[col], b2b = b2[16 + col];
    v8f cbias0, cbias1;
    for (int r = 0; r < 8; ++r) { cbias0[r] = b2a; cbias1[r] = b2b; }

    const float* efrow = ef + (size_t)i * NN;
    const float* pjf   = pj + (size_t)f * NN * EE;
    float acc0 = 0.f, acc1 = 0.f;

    for (int chunk = 0; chunk < 32; ++chunk) {
        const int j0 = chunk * 16;
        if (chunk < 31)
            __builtin_prefetch(&pjf[(size_t)(j0 + 16 + col) * EE + kh], 0, 1);

        const float efv = efrow[j0 + col];
        const float* pjrow = pjf + (size_t)(j0 + col) * EE;
        float pv[16];
        *(float4*)&pv[0]  = *(const float4*)(pjrow + kh);      // K kh..kh+3
        *(float4*)&pv[4]  = *(const float4*)(pjrow + kh + 4);  // K kh+4..kh+7
        *(float4*)&pv[8]  = *(const float4*)(pjrow + kh + 16); // K kh+16..kh+19
        *(float4*)&pv[12] = *(const float4*)(pjrow + kh + 20); // K kh+20..kh+23

        union { _Float16 h[16]; v16h v; } tv;
        for (int c = 0; c < 16; ++c)
            tv.h[c] = (_Float16)fmaxf(fmaf(efv, w1ev[c], pv[c] + pibv[c]), 0.f);

        v8f cc0 = __builtin_amdgcn_wmma_f32_16x16x32_f16(false, tv.v, false, b0,  (short)0, cbias0, false, false);
        v8f cc1 = __builtin_amdgcn_wmma_f32_16x16x32_f16(false, tv.v, false, b1v, (short)0, cbias1, false, false);
        for (int r = 0; r < 8; ++r) {
            acc0 += fmaxf(cc0[r], 0.f);
            acc1 += fmaxf(cc1[r], 0.f);
        }
    }
    // lanes n / n+16 hold rows 0-7 / 8-15 of the same column: combine.
    acc0 += __shfl_xor(acc0, 16, 32);
    acc1 += __shfl_xor(acc1, 16, 32);
    float* aggp = agg + (size_t)(f*NN + i) * EE;
    if (lane < 16) aggp[col] = acc0;
    else           aggp[16 + col] = acc1;
}

// ---------------------------------------------------------------------------
// K5: node update — h' = relu(h@nw1[:d] + agg@nw1[d:] + nb1) @ nw2 + nb2
// ---------------------------------------------------------------------------
__global__ void node_kernel(const float* __restrict__ h, int dh, int stride_h,
                            const float* __restrict__ agg,
                            const float* __restrict__ nw1, const float* __restrict__ nb1,
                            const float* __restrict__ nw2, const float* __restrict__ nb2,
                            float* __restrict__ hout)
{
    int gw = (blockIdx.x * blockDim.x + threadIdx.x) >> 5; // f*512+n
    int lane = threadIdx.x & 31;
    const float* hp = h + gw * stride_h;
    const float* ap = agg + gw * EE;
    float acc = nb1[lane];
    for (int k = 0; k < dh; ++k) acc = fmaf(hp[k], nw1[k*EE + lane], acc);
    for (int k = 0; k < EE; ++k) acc = fmaf(ap[k], nw1[(dh + k)*EE + lane], acc);
    float dn = fmaxf(acc, 0.f);
    float acc2 = nb2[lane];
    for (int k = 0; k < EE; ++k)
        acc2 = fmaf(__shfl(dn, k, 32), nw2[k*EE + lane], acc2);
    hout[gw*EE + lane] = acc2;
}

// ---------------------------------------------------------------------------
// K6: decoder + frame inversion + outputs (one wave per node)
// ---------------------------------------------------------------------------
__global__ void final_kernel(const float* __restrict__ h2, const float* __restrict__ F,
                             const float* __restrict__ x, const float* __restrict__ v,
                             const float* __restrict__ dw1, const float* __restrict__ db1,
                             const float* __restrict__ dw2, const float* __restrict__ db2,
                             const float* __restrict__ vw1, const float* __restrict__ vb1,
                             const float* __restrict__ vw2, const float* __restrict__ vb2,
                             float* __restrict__ out_h, float* __restrict__ out_x,
                             float* __restrict__ out_v)
{
    int n = (blockIdx.x * blockDim.x + threadIdx.x) >> 5; // node
    int lane = threadIdx.x & 31;
    float hsum = 0.f;
    float dxa[3] = {0,0,0}, dva[3] = {0,0,0};
    for (int f = 0; f < 8; ++f) {
        float he = h2[(size_t)(f*NN + n)*EE + lane];
        hsum += he;
        float hr = fmaxf(he, 0.f);
        float s = db1[lane];
        for (int k = 0; k < EE; ++k)
            s = fmaf(__shfl(hr, k, 32), dw1[k*EE + lane], s);
        s = fmaxf(s, 0.f);
        float sp = (lane < 6) ? db2[lane] : 0.f;
        for (int k = 0; k < EE; ++k) {
            float wv = (lane < 6) ? dw2[k*6 + lane] : 0.f;
            sp = fmaf(__shfl(s, k, 32), wv, sp);
        }
        float sp0 = __shfl(sp, 0, 32), sp1 = __shfl(sp, 1, 32), sp2 = __shfl(sp, 2, 32);
        float sp3 = __shfl(sp, 3, 32), sp4 = __shfl(sp, 4, 32), sp5 = __shfl(sp, 5, 32);
        const float* Ff = F + f*9;
        for (int i = 0; i < 3; ++i) {
            dxa[i] += Ff[i*3+0]*sp0 + Ff[i*3+1]*sp1 + Ff[i*3+2]*sp2;
            dva[i] += Ff[i*3+0]*sp3 + Ff[i*3+1]*sp4 + Ff[i*3+2]*sp5;
        }
    }
    float hm = hsum * 0.125f;
    out_h[n*EE + lane] = hm;

    float hmr = fmaxf(hm, 0.f);
    float c = vb1[lane];
    for (int k = 0; k < EE; ++k)
        c = fmaf(__shfl(hmr, k, 32), vw1[k*EE + lane], c);
    c = fmaxf(c, 0.f);
    float t = c * vw2[lane];
    for (int o = 16; o > 0; o >>= 1) t += __shfl_xor(t, o, 32);
    float coef = t + vb2[0];

    if (lane < 3) {
        float vo = v[n*3 + lane] + dva[lane] * 0.125f;
        float xo = x[n*3 + lane] + vo * coef + dxa[lane] * 0.125f;
        out_v[n*3 + lane] = vo;
        out_x[n*3 + lane] = xo;
    }
}

// ---------------------------------------------------------------------------
extern "C" void kernel_launch(void* const* d_in, const int* in_sizes, int n_in,
                              void* d_out, int out_size, void* d_ws, size_t ws_size,
                              hipStream_t stream)
{
    (void)in_sizes; (void)n_in; (void)out_size; (void)ws_size;
    const float* node_feat = (const float*)d_in[0];
    const float* edge_feat = (const float*)d_in[1];
    const float* x  = (const float*)d_in[2];
    const float* v  = (const float*)d_in[3];
    const float* e0_w1 = (const float*)d_in[4];  const float* e0_b1 = (const float*)d_in[5];
    const float* e0_w2 = (const float*)d_in[6];  const float* e0_b2 = (const float*)d_in[7];
    const float* n0_w1 = (const float*)d_in[8];  const float* n0_b1 = (const float*)d_in[9];
    const float* n0_w2 = (const float*)d_in[10]; const float* n0_b2 = (const float*)d_in[11];
    const float* e1_w1 = (const float*)d_in[12]; const float* e1_b1 = (const float*)d_in[13];
    const float* e1_w2 = (const float*)d_in[14]; const float* e1_b2 = (const float*)d_in[15];
    const float* n1_w1 = (const float*)d_in[16]; const float* n1_b1 = (const float*)d_in[17];
    const float* n1_w2 = (const float*)d_in[18]; const float* n1_b2 = (const float*)d_in[19];
    const float* dec_w1 = (const float*)d_in[20]; const float* dec_b1 = (const float*)d_in[21];
    const float* dec_w2 = (const float*)d_in[22]; const float* dec_b2 = (const float*)d_in[23];
    const float* vc_w1 = (const float*)d_in[24]; const float* vc_b1 = (const float*)d_in[25];
    const float* vc_w2 = (const float*)d_in[26]; const float* vc_b2 = (const float*)d_in[27];

    float* ws = (float*)d_ws;
    float* F      = ws;            // 72
    float* center = ws + 72;       // 3
    float* h0     = ws + 128;      // 8*512*16 = 65536
    float* h1     = ws + 65664;    // 8*512*32 = 131072
    float* h2     = ws + 196736;   // 131072
    float* pj     = ws + 327808;   // 131072
    float* pi     = ws + 458880;   // 131072
    float* agg    = ws + 589952;   // 131072  (total ~2.88 MB)

    float* out   = (float*)d_out;
    float* out_h = out;             // 512*32
    float* out_e = out + 16384;     // 512*512
    float* out_x = out + 278528;    // 512*3
    float* out_v = out + 280064;    // 512*3

    hipMemcpyAsync(out_e, edge_feat, (size_t)NN * NN * sizeof(float),
                   hipMemcpyDeviceToDevice, stream);

    frame_kernel<<<1, 32, 0, stream>>>(x, F, center);
    embed_kernel<<<16, 256, 0, stream>>>(x, v, node_feat, F, center, h0);

    dim3 egrid(64, 8);
    // ---- layer 0 (d = 14, h stride 16) ----
    prep_kernel<<<512, 256, 0, stream>>>(h0, 14, 16, e0_w1 + 32, e0_w1 + 15*32, pj, pi);
    edge_kernel<<<egrid, 256, 0, stream>>>(edge_feat, pj, pi, e0_w1, e0_b1, e0_w2, e0_b2, agg);
    node_kernel<<<512, 256, 0, stream>>>(h0, 14, 16, agg, n0_w1, n0_b1, n0_w2, n0_b2, h1);
    // ---- layer 1 (d = 32) ----
    prep_kernel<<<512, 256, 0, stream>>>(h1, 32, 32, e1_w1 + 32, e1_w1 + 33*32, pj, pi);
    edge_kernel<<<egrid, 256, 0, stream>>>(edge_feat, pj, pi, e1_w1, e1_b1, e1_w2, e1_b2, agg);
    node_kernel<<<512, 256, 0, stream>>>(h1, 32, 32, agg, n1_w1, n1_b1, n1_w2, n1_b2, h2);

    final_kernel<<<64, 256, 0, stream>>>(h2, F, x, v,
                                         dec_w1, dec_b1, dec_w2, dec_b2,
                                         vc_w1, vc_b1, vc_w2, vc_b2,
                                         out_h, out_x, out_v);
}
